// T5AttentionPrefixWrapper_29884382446381
// MI455X (gfx1250) — matrix operand order repaired and checked by reading, not compiled, not run on hardware
//
#include <hip/hip_runtime.h>
#include <math.h>

// ---------------------------------------------------------------------------
// T5 prefix attention for MI455X (gfx1250, wave32, WMMA bf16 w/ f32 accum)
// B=2 S=2048 D=1024 H=16 dh=64 P=32  -> L = 2080
// GEMMs stage W tiles into LDS via gfx1250 async loads (double-buffered).
// ---------------------------------------------------------------------------

typedef __attribute__((ext_vector_type(16))) __bf16 v16bf;
typedef __attribute__((ext_vector_type(8)))  float  v8f;
typedef __attribute__((ext_vector_type(2)))  __bf16 bf16x2;

typedef int v4i __attribute__((vector_size(16)));
typedef __attribute__((address_space(1))) v4i* gptr_v4i;   // global int4*
typedef __attribute__((address_space(3))) v4i* lptr_v4i;   // LDS int4*

#define B_  2
#define S_  2048
#define D_  1024
#define H_  16
#define P_  32
#define DH  64
#define L_  (S_ + P_)   // 2080
#define M_  (B_ * S_)   // 4096

#if __has_builtin(__builtin_amdgcn_global_load_async_to_lds_b128)
#define GEMM_ASYNC 1
#else
#define GEMM_ASYNC 0
#endif

#if __has_builtin(__builtin_amdgcn_s_wait_asynccnt)
#define WAIT_ASYNC(n) __builtin_amdgcn_s_wait_asynccnt(n)
#else
#define WAIT_ASYNC(n) asm volatile("s_wait_asynccnt %0" ::"n"(n) : "memory")
#endif

// float -> bf16, round-to-nearest-even
__device__ __forceinline__ __bf16 f2bf(float f) {
  unsigned u = __builtin_bit_cast(unsigned, f);
  u = (u + 0x7FFFu + ((u >> 16) & 1u)) >> 16;
  unsigned short s = (unsigned short)u;
  return __builtin_bit_cast(__bf16, s);
}

// WMMA 16-bit A-matrix K map: element j of lane L -> K
__device__ __forceinline__ int kmapA(int j, int half) { return (j < 8 ? j : j + 8) + half * 8; }
// WMMA 16-bit B-matrix K map
__device__ __forceinline__ int kmapB(int j, int half) { return (j < 8 ? j : j + 8) + half * 8; }

// -------------------- elementwise conversions --------------------
__global__ void k_f2bf(const float* __restrict__ in, __bf16* __restrict__ out, int n) {
  int i = blockIdx.x * blockDim.x + threadIdx.x;
  if (i < n) out[i] = f2bf(in[i]);
}

// w: [k,n] row-major fp32  ->  wt: [n,k] bf16 (transposed)
__global__ void k_transpose_f2bf(const float* __restrict__ w, __bf16* __restrict__ wt,
                                 int n, int k) {
  int i = blockIdx.x * blockDim.x + threadIdx.x;
  if (i < n * k) {
    int nn = i / k, kk = i % k;
    wt[i] = f2bf(w[kk * n + nn]);
  }
}

// prefix_k [B,H,P,dh] -> Kb rows [0,P); prefix_v -> Vt (transposed) cols [0,P)
__global__ void k_prefix(const float* __restrict__ pk, const float* __restrict__ pv,
                         __bf16* __restrict__ kb, __bf16* __restrict__ vt) {
  int i = blockIdx.x * blockDim.x + threadIdx.x;
  int total = B_ * H_ * P_ * DH;
  if (i >= total) return;
  int d  = i % DH;
  int p  = (i / DH) % P_;
  int bh = i / (DH * P_);
  kb[((long)bh * L_ + p) * DH + d] = f2bf(pk[i]);
  vt[((long)bh * DH + d) * L_ + p] = f2bf(pv[i]);
}

// -------------------- WMMA GEMM: C[M,N] = A[M,K] * W[K,N], W given as WT[N,K]
// Block = 4 waves -> 64x64 output tile; WT K-slab staged in LDS (async, 2-buf).
// mode 0: write Q  bf16 [B,H,S,dh]
// mode 1: write K  bf16 [B,H,L,dh] at row offset P
// mode 2: write Vt bf16 [B,H,dh,L] at col offset P
// mode 3: write fp32 [M,N]
__global__ void __launch_bounds__(128)
k_gemm(const __bf16* __restrict__ A, const __bf16* __restrict__ WT,
       int Mdim, int Ndim, int Kdim, int mode,
       __bf16* __restrict__ outb, float* __restrict__ outf) {
  __shared__ __attribute__((aligned(16))) __bf16 bt[2][64][32];  // 8 KiB, 2-buffered

  int lane = threadIdx.x & 31;
  int wvl  = threadIdx.x >> 5;
  int tilesN = Ndim >> 6;
  int tmb = blockIdx.x / tilesN;   // 64-row tile index
  int tn  = blockIdx.x % tilesN;   // 64-col tile index
  int rowBase = tmb * 64 + wvl * 16;
  int half = lane >> 4, lo = lane & 15;

#if GEMM_ASYNC
  // Cooperative async stage of WT[tn*64 .. +64)[kk .. kk+32) into bt[buf].
  // 64 rows x 64 bytes = 256 x 16B chunks; 128 threads x 2 chunks.
  auto stage = [&](int buf, int kk) {
    int t = threadIdx.x;
#pragma unroll
    for (int i = 0; i < 2; ++i) {
      int q = t + i * 128;                 // chunk id 0..255
      int nl = q >> 2;                     // row within tile
      int part = (q & 3) * 8;              // bf16 offset within row
      const __bf16* g = WT + (long)(tn * 64 + nl) * Kdim + kk + part;
      __bf16* l = &bt[buf][nl][part];
      __builtin_amdgcn_global_load_async_to_lds_b128(
          (gptr_v4i)(void*)const_cast<__bf16*>(g),
          (lptr_v4i)(void*)l, 0, 0);
    }
  };
#endif

  v8f acc[4] = {};
  const __bf16* arow = A + (long)(rowBase + lo) * Kdim;
  int nsteps = Kdim / 32;

#if GEMM_ASYNC
  stage(0, 0);
#endif

  for (int s = 0; s < nsteps; ++s) {
    int kk = s * 32;
#if GEMM_ASYNC
    int buf = s & 1;
    if (s + 1 < nsteps) {
      stage(buf ^ 1, kk + 32);
      WAIT_ASYNC(2);   // own ops for slab s complete (in-order)
    } else {
      WAIT_ASYNC(0);
    }
    __syncthreads();   // all waves' writes to bt[buf] visible
#endif
    v16bf af;
#pragma unroll
    for (int p = 0; p < 8; ++p) {
      bf16x2 t = *(const bf16x2*)(arow + kk + kmapA(2 * p, half));
      af[2 * p] = t.x; af[2 * p + 1] = t.y;
    }
#pragma unroll
    for (int sub = 0; sub < 4; ++sub) {
      v16bf bfrag;
#if GEMM_ASYNC
      const __bf16* wrow = &bt[buf][sub * 16 + lo][0];
#pragma unroll
      for (int p = 0; p < 8; ++p) {
        bf16x2 t = *(const bf16x2*)(wrow + kmapB(2 * p, half));
        bfrag[2 * p] = t.x; bfrag[2 * p + 1] = t.y;
      }
#else
      const __bf16* wrow = WT + (long)(tn * 64 + sub * 16 + lo) * Kdim;
#pragma unroll
      for (int p = 0; p < 8; ++p) {
        bf16x2 t = *(const bf16x2*)(wrow + kk + kmapB(2 * p, half));
        bfrag[2 * p] = t.x; bfrag[2 * p + 1] = t.y;
      }
#endif
      acc[sub] = __builtin_amdgcn_wmma_f32_16x16x32_bf16(
          false, af, false, bfrag, (short)0, acc[sub], false, false);
    }
#if GEMM_ASYNC
    __syncthreads();   // done reading bt[buf] before it is overwritten
#endif
  }

#pragma unroll
  for (int sub = 0; sub < 4; ++sub) {
#pragma unroll
    for (int r = 0; r < 8; ++r) {
      int mrow = rowBase + r + half * 8;
      int ncol = tn * 64 + sub * 16 + lo;
      float v = acc[sub][r];
      if (mode == 3) {
        outf[(long)mrow * Ndim + ncol] = v;
      } else {
        int b = mrow / S_, s = mrow % S_;
        int h = ncol / DH, d = ncol % DH;
        long bh = (long)b * H_ + h;
        if (mode == 0)      outb[(bh * S_ + s) * DH + d]        = f2bf(v);  // Q
        else if (mode == 1) outb[(bh * L_ + (P_ + s)) * DH + d] = f2bf(v);  // K
        else                outb[(bh * DH + d) * L_ + (P_ + s)] = f2bf(v);  // V^T
      }
    }
  }
}

// -------------------- flash attention --------------------
__device__ __forceinline__ float half_max(float x) {
#pragma unroll
  for (int m = 1; m < 16; m <<= 1) x = fmaxf(x, __shfl_xor(x, m, 32));
  return x;
}
__device__ __forceinline__ float half_sum(float x) {
#pragma unroll
  for (int m = 1; m < 16; m <<= 1) x += __shfl_xor(x, m, 32);
  return x;
}

// T5 bidirectional relative-position bias, NUM_BUCKETS=32 (nb=16, max_exact=8,
// max_distance=128).  8/log(128/8) = 8/log(16).
__device__ __forceinline__ float t5_bias(int rel, int h, const float* __restrict__ rt) {
  int ret = rel > 0 ? 16 : 0;
  int n = rel < 0 ? -rel : rel;
  int b;
  if (n < 8) {
    b = n;
  } else {
    int vl = 8 + (int)(__logf((float)n * 0.125f) * 2.8853900817779268f);
    b = vl < 15 ? vl : 15;
  }
  return rt[(ret + b) * H_ + h];
}

__global__ void __launch_bounds__(128)
k_flash(const __bf16* __restrict__ Qb, const __bf16* __restrict__ Kb,
        const __bf16* __restrict__ Vt, const float* __restrict__ rt,
        const float* __restrict__ mask, __bf16* __restrict__ Ob) {
  __shared__ __attribute__((aligned(16))) __bf16 plds[4][16 * 32];  // per-wave P tile

  int lane = threadIdx.x & 31;
  int wvl  = threadIdx.x >> 5;
  int wv   = blockIdx.x * 4 + wvl;
  int qtiles = S_ / 16;
  int bh = wv / qtiles;
  int qt = wv % qtiles;
  int b = bh / H_, h = bh % H_;
  int half = lane >> 4, lo = lane & 15;

  // Q tile: 16 rows x dh=64, as two 16x32 A fragments (resident)
  const __bf16* qbase = Qb + ((long)bh * S_ + qt * 16 + lo) * DH;
  v16bf qa[2];
#pragma unroll
  for (int t = 0; t < 2; ++t)
#pragma unroll
    for (int p = 0; p < 8; ++p) {
      bf16x2 v = *(const bf16x2*)(qbase + t * 32 + kmapA(2 * p, half));
      qa[t][2 * p] = v.x; qa[t][2 * p + 1] = v.y;
    }

  v8f Oacc[4] = {};
  float run_m[8], run_l[8];
#pragma unroll
  for (int r = 0; r < 8; ++r) { run_m[r] = -1e30f; run_l[r] = 0.f; }

  const __bf16* kbh = Kb + (long)bh * L_ * DH;
  const __bf16* vbh = Vt + (long)bh * DH * L_;

  for (int c = 0; c < L_ / 32; ++c) {        // 65 chunks of 32 kv positions
    // ---- scores: q(16x64) @ k^T(64x32) via 4 WMMAs ----
    v8f sc[2] = {};
#pragma unroll
    for (int nt = 0; nt < 2; ++nt) {
      const __bf16* krow = kbh + (long)(c * 32 + nt * 16 + lo) * DH;
#pragma unroll
      for (int kk = 0; kk < 2; ++kk) {
        v16bf bfrag;
#pragma unroll
        for (int p = 0; p < 8; ++p) {
          bf16x2 v = *(const bf16x2*)(krow + kk * 32 + kmapB(2 * p, half));
          bfrag[2 * p] = v.x; bfrag[2 * p + 1] = v.y;
        }
        sc[nt] = __builtin_amdgcn_wmma_f32_16x16x32_bf16(
            false, qa[kk], false, bfrag, (short)0, sc[nt], false, false);
      }
    }
    // ---- bias + mask ----
#pragma unroll
    for (int nt = 0; nt < 2; ++nt) {
      int kg = c * 32 + nt * 16 + lo;
      float mval = (kg >= P_) ? mask[(long)b * S_ + (kg - P_)] : 0.f;
#pragma unroll
      for (int r = 0; r < 8; ++r) {
        int qg = qt * 16 + r + half * 8;
        sc[nt][r] += t5_bias(kg - (qg + P_), h, rt) + mval;
      }
    }
    // ---- online softmax; stash P (bf16) in LDS ----
    float scale8[8];
#pragma unroll
    for (int r = 0; r < 8; ++r) {
      float mc = half_max(fmaxf(sc[0][r], sc[1][r]));
      float mn = fmaxf(run_m[r], mc);
      float sl = __expf(run_m[r] - mn);
      run_m[r] = mn;
      float p0 = __expf(sc[0][r] - mn);
      float p1 = __expf(sc[1][r] - mn);
      run_l[r] = run_l[r] * sl + half_sum(p0 + p1);
      scale8[r] = sl;
      int row = r + half * 8;
      plds[wvl][row * 32 + lo]      = f2bf(p0);
      plds[wvl][row * 32 + 16 + lo] = f2bf(p1);
    }
#pragma unroll
    for (int t = 0; t < 4; ++t)
#pragma unroll
      for (int r = 0; r < 8; ++r) Oacc[t][r] *= scale8[r];

    // reload P in A-fragment layout (C-layout -> A-layout via LDS; wave-order)
    v16bf pf;
#pragma unroll
    for (int p = 0; p < 8; ++p) {
      bf16x2 v = *(const bf16x2*)(&plds[wvl][lo * 32 + kmapA(2 * p, half)]);
      pf[2 * p] = v.x; pf[2 * p + 1] = v.y;
    }
    // ---- O += P(16x32) @ V(32x64) via 4 WMMAs ----
#pragma unroll
    for (int t = 0; t < 4; ++t) {
      const __bf16* vrow = vbh + (long)(t * 16 + lo) * L_ + c * 32;
      v16bf vf;
#pragma unroll
      for (int p = 0; p < 8; ++p) {
        bf16x2 v = *(const bf16x2*)(vrow + kmapB(2 * p, half));
        vf[2 * p] = v.x; vf[2 * p + 1] = v.y;
      }
      Oacc[t] = __builtin_amdgcn_wmma_f32_16x16x32_bf16(
          false, pf, false, vf, (short)0, Oacc[t], false, false);
    }
  }
  // ---- normalize + write attn output bf16 [B,S,D] ----
#pragma unroll
  for (int r = 0; r < 8; ++r) {
    float inv = 1.0f / run_l[r];
    int s = qt * 16 + r + half * 8;
#pragma unroll
    for (int t = 0; t < 4; ++t)
      Ob[((long)b * S_ + s) * D_ + h * DH + t * 16 + lo] = f2bf(Oacc[t][r] * inv);
  }
}

// -------------------- host orchestration --------------------
extern "C" void kernel_launch(void* const* d_in, const int* in_sizes, int n_in,
                              void* d_out, int out_size, void* d_ws, size_t ws_size,
                              hipStream_t stream) {
  const float* X    = (const float*)d_in[0];
  const float* mask = (const float*)d_in[1];
  const float* pk   = (const float*)d_in[2];
  const float* pv   = (const float*)d_in[3];
  const float* Wq   = (const float*)d_in[4];
  const float* Wk   = (const float*)d_in[5];
  const float* Wv   = (const float*)d_in[6];
  const float* Wo   = (const float*)d_in[7];
  const float* rt   = (const float*)d_in[8];
  float* out = (float*)d_out;

  char* ws = (char*)d_ws;
  size_t off = 0;
  auto alloc = [&](size_t bytes) -> char* {
    char* p = ws + off;
    off = (off + bytes + 255) & ~(size_t)255;
    return p;
  };
  __bf16* Xbf = (__bf16*)alloc((size_t)M_ * D_ * 2);          // 8 MiB
  __bf16* WqT = (__bf16*)alloc((size_t)D_ * D_ * 2);          // 2 MiB
  __bf16* WkT = (__bf16*)alloc((size_t)D_ * D_ * 2);
  __bf16* WvT = (__bf16*)alloc((size_t)D_ * D_ * 2);
  __bf16* WoT = (__bf16*)alloc((size_t)D_ * D_ * 2);
  __bf16* Qbf = (__bf16*)alloc((size_t)B_ * H_ * S_ * DH * 2);  // 8 MiB
  __bf16* Kbf = (__bf16*)alloc((size_t)B_ * H_ * L_ * DH * 2);  // 8.1 MiB
  __bf16* Vtb = (__bf16*)alloc((size_t)B_ * H_ * DH * L_ * 2);  // 8.1 MiB
  __bf16* Obf = (__bf16*)alloc((size_t)M_ * D_ * 2);            // 8 MiB
  (void)ws_size; (void)in_sizes; (void)n_in; (void)out_size;

  k_f2bf<<<(M_ * D_ + 255) / 256, 256, 0, stream>>>(X, Xbf, M_ * D_);
  k_transpose_f2bf<<<(D_ * D_ + 255) / 256, 256, 0, stream>>>(Wq, WqT, D_, D_);
  k_transpose_f2bf<<<(D_ * D_ + 255) / 256, 256, 0, stream>>>(Wk, WkT, D_, D_);
  k_transpose_f2bf<<<(D_ * D_ + 255) / 256, 256, 0, stream>>>(Wv, WvT, D_, D_);
  k_transpose_f2bf<<<(D_ * D_ + 255) / 256, 256, 0, stream>>>(Wo, WoT, D_, D_);
  k_prefix<<<(B_ * H_ * P_ * DH + 255) / 256, 256, 0, stream>>>(pk, pv, Kbf, Vtb);

  int gemmBlocks = (M_ / 64) * (D_ / 64);  // 1024 blocks, 4 waves each (64x64 tile)
  k_gemm<<<gemmBlocks, 128, 0, stream>>>(Xbf, WqT, M_, D_, D_, 0, Qbf, nullptr);
  k_gemm<<<gemmBlocks, 128, 0, stream>>>(Xbf, WkT, M_, D_, D_, 1, Kbf, nullptr);
  k_gemm<<<gemmBlocks, 128, 0, stream>>>(Xbf, WvT, M_, D_, D_, 2, Vtb, nullptr);

  int fBlocks = (B_ * H_ * (S_ / 16)) / 4;     // 1024 blocks
  k_flash<<<fBlocks, 128, 0, stream>>>(Qbf, Kbf, Vtb, rt, mask, Obf);

  k_gemm<<<gemmBlocks, 128, 0, stream>>>(Obf, WoT, M_, D_, D_, 3, nullptr, out);
}
